// LSTMCell_13159779795584
// MI455X (gfx1250) — matrix operand order repaired and checked by reading, not compiled, not run on hardware
//
#include <hip/hip_runtime.h>
#include <hip/hip_bf16.h>

#define B_ 64
#define T_ 512
#define I_ 512
#define H_ 1024
#define M_ (B_ * T_)   // 32768 rows for input projection GEMM

typedef __attribute__((ext_vector_type(16))) __bf16 v16bf;
typedef __attribute__((ext_vector_type(8)))  float  v8f;
typedef __attribute__((ext_vector_type(8)))  unsigned short us8;
typedef __attribute__((ext_vector_type(16))) unsigned short us16;

// ---------- workspace layout (bytes) ----------
// pre    : float [4][M_][H_]           = 512 MiB  (gate pre-activations x@W + b)
// WhbfT  : bf16 [4][H_][H_]  ([n][k])  =   8 MiB  (recurrent weights, transposed)
// hbf    : bf16 [B_][H_]               = 128 KiB  (carry h)
// cnt    : int (padded)                            (global barrier counter)
// xbf    : bf16 [M_][I_]               =  32 MiB  (x converted once)
// WibfT  : bf16 [4][H_][I_] ([n][k])   =   4 MiB  (input weights, transposed)
static constexpr size_t PRE_OFF  = 0;
static constexpr size_t WHBF_OFF = PRE_OFF + (size_t)4 * M_ * H_ * sizeof(float);
static constexpr size_t HBF_OFF  = WHBF_OFF + (size_t)4 * H_ * H_ * sizeof(unsigned short);
static constexpr size_t CNT_OFF  = HBF_OFF + (size_t)B_ * H_ * sizeof(unsigned short);
static constexpr size_t XBF_OFF  = CNT_OFF + 256;
static constexpr size_t WIBF_OFF = XBF_OFF + (size_t)M_ * I_ * sizeof(unsigned short);

__device__ __forceinline__ unsigned short f2bf(float f) {
    union { float f; unsigned u; } v; v.f = f;
    unsigned r = v.u + 0x7FFFu + ((v.u >> 16) & 1u);   // round-to-nearest-even
    return (unsigned short)(r >> 16);
}

__device__ __forceinline__ v8f vzero8() {
    v8f v;
#pragma unroll
    for (int i = 0; i < 8; ++i) v[i] = 0.0f;
    return v;
}

// A-fragment: two 16-byte chunks (K 16h+0..7 / 16h+8..15 for lane half h; contiguous k order)
__device__ __forceinline__ v16bf frag_split(const unsigned short* p0, const unsigned short* p1) {
    us8 a = *(const us8*)p0;
    us8 b = *(const us8*)p1;
    us16 t;
#pragma unroll
    for (int i = 0; i < 8; ++i) { t[i] = a[i]; t[i + 8] = b[i]; }
    return __builtin_bit_cast(v16bf, t);
}

// B-fragment: one contiguous 32-byte chunk (16 K values for this lane's column)
__device__ __forceinline__ v16bf frag_contig(const unsigned short* p) {
    us16 t = *(const us16*)p;
    return __builtin_bit_cast(v16bf, t);
}

__device__ __forceinline__ float sigmoid_f(float x) {
    return 1.0f / (1.0f + __expf(-x));
}
__device__ __forceinline__ float tanh_f(float x) {
    float e = __expf(2.0f * x);
    return (e - 1.0f) / (e + 1.0f);
}

// CDNA5 async global->LDS 16B copy (ASYNCcnt-tracked DMA into LDS).
__device__ __forceinline__ void async_cp16(unsigned ldsoff, const void* base, unsigned off) {
    asm volatile("global_load_async_to_lds_b128 %0, %1, %2"
                 :: "v"(ldsoff), "v"(off), "s"((unsigned long long)base)
                 : "memory");
}
__device__ __forceinline__ void async_wait0() { asm volatile("s_wait_asynccnt 0x0" ::: "memory"); }
__device__ __forceinline__ void async_wait3() { asm volatile("s_wait_asynccnt 0x3" ::: "memory"); }
__device__ __forceinline__ void async_wait4() { asm volatile("s_wait_asynccnt 0x4" ::: "memory"); }

// ---------------------------------------------------------------------------
// Prep: convert everything to bf16 once.
//  xbf[m][k]      = bf16(x[m][k])
//  WibfT[g][n][k] = bf16(W_i{g}[k][n])     (k over I_)
//  WhbfT[g][n][k] = bf16(W_h{g}[k][n])     (k over H_)
//  hbf = 0, cnt = 0
// ---------------------------------------------------------------------------
__global__ void lstm_prep(const float* __restrict__ x,
                          const float* __restrict__ Wi0, const float* __restrict__ Wi1,
                          const float* __restrict__ Wi2, const float* __restrict__ Wi3,
                          const float* __restrict__ Wh0, const float* __restrict__ Wh1,
                          const float* __restrict__ Wh2, const float* __restrict__ Wh3,
                          unsigned short* __restrict__ xbf,
                          unsigned short* __restrict__ WibfT,
                          unsigned short* __restrict__ WhbfT,
                          unsigned short* __restrict__ hbf,
                          int* __restrict__ cnt) {
    const size_t NT  = (size_t)gridDim.x * blockDim.x;
    const size_t idx = (size_t)blockIdx.x * blockDim.x + threadIdx.x;

    const size_t N1 = (size_t)M_ * I_;        // x elements
    for (size_t e = idx; e < N1; e += NT) xbf[e] = f2bf(x[e]);

    const size_t NWI = (size_t)H_ * I_;
    for (size_t e = idx; e < 4 * NWI; e += NT) {
        size_t g = e / NWI, r = e - g * NWI;
        size_t n = r / I_,  k = r - n * I_;
        const float* W = (g == 0) ? Wi0 : (g == 1) ? Wi1 : (g == 2) ? Wi2 : Wi3;
        WibfT[e] = f2bf(W[k * H_ + n]);
    }

    const size_t NWH = (size_t)H_ * H_;
    for (size_t e = idx; e < 4 * NWH; e += NT) {
        size_t g = e / NWH, r = e - g * NWH;
        size_t n = r / H_,  k = r - n * H_;
        const float* W = (g == 0) ? Wh0 : (g == 1) ? Wh1 : (g == 2) ? Wh2 : Wh3;
        WhbfT[e] = f2bf(W[k * H_ + n]);
    }

    if (idx < (size_t)B_ * H_) hbf[idx] = 0;   // bf16 zero == 0x0000
    if (idx == 0) *cnt = 0;
}

// ---------------------------------------------------------------------------
// Phase 1: input projections.  pre[g] = x @ W_i{g} + b{g}   (bf16 WMMA, f32 acc)
// grid = (M_/128, 32); blockIdx.y -> (gate = y>>3, n0 = (y&7)*128)
// block = 256 threads = 8 waves; block tile 128x128; wave tile 32x64 (2x4 WMMA).
// All staging = double-buffered async global->LDS DMA overlapped with WMMA.
// ---------------------------------------------------------------------------
__global__ void __launch_bounds__(256)
lstm_xproj(const unsigned short* __restrict__ xbf,
           const unsigned short* __restrict__ WibfT,
           const float* __restrict__ bi0, const float* __restrict__ bi1,
           const float* __restrict__ bi2, const float* __restrict__ bi3,
           float* __restrict__ pre) {
    const int gate = blockIdx.y >> 3;
    const int n0   = (blockIdx.y & 7) * 128;
    const int m0   = blockIdx.x * 128;

    const float* bias = (gate == 0) ? bi0 : (gate == 1) ? bi1 : (gate == 2) ? bi2 : bi3;
    float* out = pre + (size_t)gate * M_ * H_;

    __shared__ alignas(32) unsigned short lA[2][128 * 32];   // [m][k] tiles, bf16
    __shared__ alignas(32) unsigned short lB[2][128 * 32];   // [n][k] tiles, bf16

    const unsigned lA_off[2] = { (unsigned)(unsigned long long)(&lA[0][0]),
                                 (unsigned)(unsigned long long)(&lA[1][0]) };
    const unsigned lB_off[2] = { (unsigned)(unsigned long long)(&lB[0][0]),
                                 (unsigned)(unsigned long long)(&lB[1][0]) };

    const int tid  = threadIdx.x;
    const int lane = tid & 31;
    const int w    = tid >> 5;
    const int wr   = w >> 1;     // 0..3  (wave row: 32 rows each)
    const int wc   = w & 1;      // 0..1  (wave col: 64 cols each)

    // per-thread staging coordinates: row (0..127) + 16-element half of the K chunk
    const int srow = tid >> 1;
    const int sch  = (tid & 1) * 16;
    const unsigned aLoc = (unsigned)((srow * 32 + sch) * 2);
    const unsigned bLoc = aLoc;

    v8f acc[2][4];
#pragma unroll
    for (int i = 0; i < 2; ++i)
#pragma unroll
        for (int j = 0; j < 4; ++j) acc[i][j] = vzero8();

    auto issue = [&](int buf, int k0) {
        unsigned ga = (unsigned)(((size_t)(m0 + srow) * I_ + k0 + sch) * 2);
        async_cp16(lA_off[buf] + aLoc,      xbf, ga);
        async_cp16(lA_off[buf] + aLoc + 16, xbf, ga + 16);
        unsigned gb = (unsigned)(((size_t)gate * H_ * I_ +
                                  (size_t)(n0 + srow) * I_ + k0 + sch) * 2);
        async_cp16(lB_off[buf] + bLoc,      WibfT, gb);
        async_cp16(lB_off[buf] + bLoc + 16, WibfT, gb + 16);
    };

    constexpr int NK = I_ / 32;   // 16 chunks
    issue(0, 0);
    for (int kc = 0; kc < NK; ++kc) {
        const int cur = kc & 1;
        if (kc + 1 < NK) { issue(cur ^ 1, (kc + 1) * 32); async_wait4(); }
        else             { async_wait0(); }
        __syncthreads();

        const int rsel = lane & 15;
        const int kg   = lane >> 4;
        v16bf afrag[2], bfrag[4];
#pragma unroll
        for (int i = 0; i < 2; ++i) {
            const unsigned short* base = &lA[cur][(wr * 32 + i * 16 + rsel) * 32];
            afrag[i] = frag_split(base + kg * 16, base + kg * 16 + 8);
        }
#pragma unroll
        for (int j = 0; j < 4; ++j) {
            bfrag[j] = frag_contig(&lB[cur][(wc * 64 + j * 16 + rsel) * 32 + kg * 16]);
        }
#pragma unroll
        for (int i = 0; i < 2; ++i)
#pragma unroll
            for (int j = 0; j < 4; ++j)
                acc[i][j] = __builtin_amdgcn_wmma_f32_16x16x32_bf16(
                    false, afrag[i], false, bfrag[j], (short)0, acc[i][j], false, false);
        __syncthreads();
    }

    // epilogue: C layout per ISA (VGPR r -> M=r (+8 for lanes>=16), N = lane&15)
    {
        const int rsel = lane & 15;
        const int hi   = lane >> 4;
#pragma unroll
        for (int i = 0; i < 2; ++i)
#pragma unroll
            for (int j = 0; j < 4; ++j)
#pragma unroll
                for (int r = 0; r < 8; ++r) {
                    int m = m0 + wr * 32 + i * 16 + r + hi * 8;
                    int n = n0 + wc * 64 + j * 16 + rsel;
                    out[(size_t)m * H_ + n] = acc[i][j][r] + bias[n];
                }
    }
}

// ---------------------------------------------------------------------------
// Phase 2: persistent recurrence.  32 blocks; block b owns h-cols [32b,32b+32).
// Per step: async-prefetch preact tile into LDS, double-buffered async staging
// for the 4 gate GEMMs (M=64, K=1024, N=32) via bf16 WMMA, elementwise cell
// update, then a device-wide arrive/spin barrier.
// ---------------------------------------------------------------------------
__global__ void __launch_bounds__(256)
lstm_recur(const float* __restrict__ pre,
           const unsigned short* __restrict__ WhbfT,  // [g][n][k]
           unsigned short* __restrict__ hbf,
           float* __restrict__ out,
           int* __restrict__ cnt) {
    const int n0   = blockIdx.x * 32;
    const int tid  = threadIdx.x;
    const int lane = tid & 31;
    const int w    = tid >> 5;
    const int g    = w >> 1;            // gate 0..3 (i,f,o,g)
    const int nh   = (w & 1) * 16;      // column half within the 32-col slice

    __shared__ alignas(32) unsigned short lA[2][64 * 32];    // h tiles        8 KiB
    __shared__ alignas(32) unsigned short lB[2][4][32 * 32]; // W tiles       16 KiB
    __shared__ alignas(16) float pbuf[4][64][32];            // preact tiles  32 KiB
    __shared__ float cbuf[64][32];                           // cell state     8 KiB

    const unsigned lA_off[2] = { (unsigned)(unsigned long long)(&lA[0][0]),
                                 (unsigned)(unsigned long long)(&lA[1][0]) };
    const unsigned lB_off[2] = { (unsigned)(unsigned long long)(&lB[0][0][0]),
                                 (unsigned)(unsigned long long)(&lB[1][0][0]) };
    const unsigned pb_off    =   (unsigned)(unsigned long long)(&pbuf[0][0][0]);

    for (int e = tid; e < 64 * 32; e += 256) cbuf[e >> 5][e & 31] = 0.0f;
    __syncthreads();

    const size_t SEQ = (size_t)B_ * T_ * H_;

    // staging coordinates
    const int arow = tid >> 2;          // 0..63  (A: h rows)
    const int ac4  = (tid & 3) * 8;     // 0,8,16,24
    const unsigned aLoc = (unsigned)((arow * 32 + ac4) * 2);
    const int bg  = tid >> 6;           // 0..3   (B: gate)
    const int bn  = (tid >> 1) & 31;    // 0..31  (B: n row)
    const int bkh = (tid & 1) * 16;     // 0/16   (B: k half)
    const unsigned bLoc = (unsigned)((bg * 1024 + bn * 32 + bkh) * 2);

    auto issue = [&](int buf, int k0) {
        async_cp16(lA_off[buf] + aLoc, hbf, (unsigned)((arow * H_ + k0 + ac4) * 2));
        unsigned goff = (unsigned)(((size_t)bg * H_ * H_ +
                                    (size_t)(n0 + bn) * H_ + k0 + bkh) * 2);
        async_cp16(lB_off[buf] + bLoc,      WhbfT, goff);
        async_cp16(lB_off[buf] + bLoc + 16, WhbfT, goff + 16);
    };

    for (int t = 0; t < T_; ++t) {
        // async-prefetch preact tile [4][64][32] f32 (lands during the GEMM)
        {
            const int pg = tid >> 6;
            const int pm = tid & 63;
            unsigned gbase = (unsigned)(((size_t)pg * M_ * H_ +
                                         ((size_t)pm * T_ + t) * H_ + n0) * sizeof(float));
            unsigned lbase = pb_off + (unsigned)((pg * 2048 + pm * 32) * sizeof(float));
#pragma unroll
            for (int j = 0; j < 8; ++j)
                async_cp16(lbase + j * 16, pre, gbase + j * 16);
        }

        v8f acc[4];
#pragma unroll
        for (int mt = 0; mt < 4; ++mt) acc[mt] = vzero8();

        constexpr int NK = H_ / 32;   // 32 chunks
        issue(0, 0);
        for (int kc = 0; kc < NK; ++kc) {
            const int cur = kc & 1;
            if (kc + 1 < NK) { issue(cur ^ 1, (kc + 1) * 32); async_wait3(); }
            else             { async_wait0(); }
            __syncthreads();

            const int rsel = lane & 15;
            const int kg   = lane >> 4;
            v16bf bf = frag_contig(&lB[cur][g][(nh + rsel) * 32 + kg * 16]);
#pragma unroll
            for (int mt = 0; mt < 4; ++mt) {
                const unsigned short* base = &lA[cur][(mt * 16 + rsel) * 32];
                v16bf af = frag_split(base + kg * 16, base + kg * 16 + 8);
                acc[mt] = __builtin_amdgcn_wmma_f32_16x16x32_bf16(
                    false, af, false, bf, (short)0, acc[mt], false, false);
            }
            __syncthreads();
        }

        // gate pre-activation: pbuf += h@Wh  (bias already folded into pre)
        {
            const int rsel = lane & 15;
            const int hi   = lane >> 4;
#pragma unroll
            for (int mt = 0; mt < 4; ++mt)
#pragma unroll
                for (int r = 0; r < 8; ++r) {
                    int m = mt * 16 + r + hi * 8;   // batch index
                    int n = nh + rsel;              // local col 0..31
                    pbuf[g][m][n] += acc[mt][r];
                }
        }
        __syncthreads();

        // elementwise LSTM cell update
#pragma unroll
        for (int i = 0; i < 8; ++i) {
            int e = tid * 8 + i;
            int m = e >> 5, n = e & 31;
            float it = sigmoid_f(pbuf[0][m][n]);
            float ft = sigmoid_f(pbuf[1][m][n]);
            float ot = sigmoid_f(pbuf[2][m][n]);
            float gt = tanh_f(pbuf[3][m][n]);
            float c  = ft * cbuf[m][n] + it * gt;
            float h  = ot * tanh_f(c);
            cbuf[m][n] = c;
            out[((size_t)m * T_ + t) * H_ + n0 + n] = h;            // hidden_seq
            hbf[(size_t)m * H_ + n0 + n] = f2bf(h);                 // carry (bf16)
            if (t == T_ - 1) {
                out[SEQ + (size_t)m * H_ + n0 + n] = h;             // h_T
                out[SEQ + (size_t)B_ * H_ + (size_t)m * H_ + n0 + n] = c;  // c_T
            }
        }
        __syncthreads();
        __threadfence();

        // device-wide barrier: all 32 blocks must publish h before next step
        if (tid == 0) {
            atomicAdd(cnt, 1);
            while (__hip_atomic_load(cnt, __ATOMIC_RELAXED, __HIP_MEMORY_SCOPE_AGENT)
                   < 32 * (t + 1)) {
                __builtin_amdgcn_s_sleep(1);
            }
        }
        __syncthreads();
    }
}

// ---------------------------------------------------------------------------
extern "C" void kernel_launch(void* const* d_in, const int* in_sizes, int n_in,
                              void* d_out, int out_size, void* d_ws, size_t ws_size,
                              hipStream_t stream) {
    (void)in_sizes; (void)n_in; (void)out_size; (void)ws_size;

    const float* x    = (const float*)d_in[0];
    const float* W_ii = (const float*)d_in[1];
    const float* W_hi = (const float*)d_in[2];
    const float* b_i  = (const float*)d_in[3];
    const float* W_if = (const float*)d_in[4];
    const float* W_hf = (const float*)d_in[5];
    const float* b_f  = (const float*)d_in[6];
    const float* W_io = (const float*)d_in[7];
    const float* W_ho = (const float*)d_in[8];
    const float* b_o  = (const float*)d_in[9];
    const float* W_ig = (const float*)d_in[10];
    const float* W_hg = (const float*)d_in[11];
    const float* b_g  = (const float*)d_in[12];

    char* ws = (char*)d_ws;
    float*          pre   = (float*)(ws + PRE_OFF);
    unsigned short* WhbfT = (unsigned short*)(ws + WHBF_OFF);
    unsigned short* hbf   = (unsigned short*)(ws + HBF_OFF);
    int*            cnt   = (int*)(ws + CNT_OFF);
    unsigned short* xbf   = (unsigned short*)(ws + XBF_OFF);
    unsigned short* WibfT = (unsigned short*)(ws + WIBF_OFF);
    float*          out   = (float*)d_out;

    lstm_prep<<<4096, 256, 0, stream>>>(x,
        W_ii, W_if, W_io, W_ig,
        W_hi, W_hf, W_ho, W_hg,
        xbf, WibfT, WhbfT, hbf, cnt);

    lstm_xproj<<<dim3(M_ / 128, 32), 256, 0, stream>>>(
        xbf, WibfT, b_i, b_f, b_o, b_g, pre);

    lstm_recur<<<32, 256, 0, stream>>>(pre, WhbfT, hbf, out, cnt);
}
